// SPALWinNNModel_36009005810281
// MI455X (gfx1250) — matrix operand order repaired and checked by reading, not compiled, not run on hardware
//
#include <hip/hip_runtime.h>
#include <hip/hip_bf16.h>

typedef __bf16 bf16_t;
typedef __attribute__((ext_vector_type(16))) __bf16 v16bf;
typedef __attribute__((ext_vector_type(8)))  __bf16 v8bf;
typedef __attribute__((ext_vector_type(8)))  float  v8f;

#define HH   56
#define WW   56
#define CC   448
#define CPAD 456          // row pad: 912B rows, 16B aligned, bank-skewed
#define NPIX 3136         // 56*56
#define NB   100          // memory bank images
#define KIM  20
#define FWIN 2
#define BIGF 1e30f
#define OUTHW 224
#define NOUT  (2 * OUTHW * OUTHW)
#define DCH   1404928ull  // 3136*448

// ---------------- stage 0: zero accumulators ----------------
__global__ void zero_acc_kernel(float* ws) {
    for (int i = threadIdx.x; i < 352; i += blockDim.x) ws[i] = 0.0f;
}

// ---------------- stage 0b: transpose (B,C,H,W) -> (B,HW,C), f32 + bf16 ----
__global__ __launch_bounds__(256) void transpose_kernel(
    const float* __restrict__ emb, float* __restrict__ et, bf16_t* __restrict__ ebf) {
    __shared__ float tile[16][17];
    const int b   = blockIdx.z;
    const int hw0 = blockIdx.x * 16;
    const int c0  = blockIdx.y * 16;
    const int tx  = threadIdx.x & 15;
    const int ty  = threadIdx.x >> 4;
    tile[ty][tx] = emb[((size_t)b * CC + c0 + ty) * NPIX + hw0 + tx];
    __syncthreads();
    const float v = tile[tx][ty];
    const size_t o = ((size_t)b * NPIX + hw0 + ty) * CC + c0 + tx;
    et[o]  = v;
    ebf[o] = (bf16_t)v;
}

// ---------------- stage 0c: per-pixel e2 and per-image e2 sum --------------
__global__ void e2_kernel(const float* __restrict__ et, float* __restrict__ e2p,
                          float* __restrict__ e2img) {
    const int i = blockIdx.x * blockDim.x + threadIdx.x;
    if (i >= 2 * NPIX) return;
    const float* p = et + (size_t)i * CC;
    float s = 0.f;
    for (int j = 0; j < CC; ++j) s += p[j] * p[j];
    e2p[i] = s;
    atomicAdd(&e2img[i / NPIX], s);
}

// ---------------- stage 1: streaming image-level dots + ||m||^2 ------------
__global__ __launch_bounds__(256) void imgdot_kernel(
    const float* __restrict__ mem, const float* __restrict__ et,
    float* __restrict__ dotw, float* __restrict__ m2n) {
    const int n   = blockIdx.y;
    const int z   = blockIdx.x;             // 8 chunks
    const int tid = threadIdx.x;
    const size_t chunk = DCH / 8;           // 175616, divisible by 256
    const size_t base  = (size_t)z * chunk;
    const float* m = mem + (size_t)n * DCH;
    float d0 = 0.f, d1 = 0.f, s2 = 0.f;
    for (size_t i = base + tid; i < base + chunk; i += 256) {
        const float mv = m[i];
        s2 += mv * mv;
        d0 += mv * et[i];
        d1 += mv * et[DCH + i];
    }
    for (int off = 16; off > 0; off >>= 1) {
        d0 += __shfl_down(d0, off, 32);
        d1 += __shfl_down(d1, off, 32);
        s2 += __shfl_down(s2, off, 32);
    }
    __shared__ float sd0[8], sd1[8], ss2[8];
    const int wave = tid >> 5, lane = tid & 31;
    if (lane == 0) { sd0[wave] = d0; sd1[wave] = d1; ss2[wave] = s2; }
    __syncthreads();
    if (tid == 0) {
        float a = 0.f, bb = 0.f, c = 0.f;
        for (int w = 0; w < 8; ++w) { a += sd0[w]; bb += sd1[w]; c += ss2[w]; }
        atomicAdd(&dotw[n], a);
        atomicAdd(&dotw[NB + n], bb);
        atomicAdd(&m2n[n], c);
    }
}

// ---------------- stage 2: distances, top-20, pred_score ------------------
__global__ void topk_kernel(const float* __restrict__ dotw, const float* __restrict__ m2n,
                            const float* __restrict__ e2img, float* __restrict__ out,
                            int* __restrict__ tidx) {
    const int b = blockIdx.x;
    __shared__ float dist[NB];
    for (int n = threadIdx.x; n < NB; n += 32) {
        const float d2 = e2img[b] + m2n[n] - 2.0f * dotw[b * NB + n];
        dist[n] = sqrtf(fmaxf(d2, 0.0f));
    }
    __syncthreads();
    if (threadIdx.x == 0) {
        bool used[NB];
        for (int n = 0; n < NB; ++n) used[n] = false;
        float s = 0.f;
        for (int k = 0; k < KIM; ++k) {
            float bv = BIGF; int bi = 0;
            for (int n = 0; n < NB; ++n)
                if (!used[n] && dist[n] < bv) { bv = dist[n]; bi = n; }
            used[bi] = true;
            tidx[b * KIM + k] = bi;
            s += bv;
        }
        out[b] = s / (float)KIM;
    }
}

// ---------------- stage 3: WMMA windowed NN search -------------------------
// block = 128 threads (4 waves). Each block: 4x4 query tile, 8x8 candidate halo.
// Wave w owns candidate tile w (16 candidates). K=448 in 14 wmma bf16 steps.
__global__ __launch_bounds__(128) void winnn_kernel(
    const float* __restrict__ mem, const bf16_t* __restrict__ ebf,
    const float* __restrict__ e2p, const int* __restrict__ tidx,
    float* __restrict__ patch) {
    __shared__ bf16_t qs[16 * CPAD];
    __shared__ bf16_t cs[64 * CPAD];
    __shared__ float  m2s[64];
    __shared__ float  e2q[16];
    __shared__ float  wmin[4][16];

    const int b    = blockIdx.y;
    const int h0   = (blockIdx.x / 14) * 4;
    const int w0   = (blockIdx.x % 14) * 4;
    const int tid  = threadIdx.x;
    const int wave = tid >> 5;
    const int lane = tid & 31;

    // queries (already bf16 in workspace) -> LDS
    for (int i = tid; i < 16 * CC; i += 128) {
        const int q = i / CC, j = i % CC;
        const int qh = h0 + (q >> 2), qw = w0 + (q & 3);
        qs[q * CPAD + j] = ebf[((size_t)b * NPIX + qh * WW + qw) * CC + j];
    }
    if (tid < 16) {
        const int qh = h0 + (tid >> 2), qw = w0 + (tid & 3);
        e2q[tid] = e2p[b * NPIX + qh * WW + qw];
    }

    float minv[8];
    for (int r = 0; r < 8; ++r) minv[r] = BIGF;

    const int  cl   = lane & 15;                 // A row / B column / D column
    const int  cid  = wave * 16 + cl;            // candidate id this lane scores
    const int  chh  = h0 - FWIN + (cid >> 3);
    const int  cww  = w0 - FWIN + (cid & 7);
    const bool cin  = (chh >= 0 && chh < HH && cww >= 0 && cww < WW);

    for (int k = 0; k < KIM; ++k) {
        const int idx = tidx[b * KIM + k];
        __syncthreads();  // WAR on cs/m2s vs previous iteration
        // cooperative f32->bf16 load of 8x8 halo + ||m||^2 per candidate
        {
            const int grp = lane >> 3, l8 = lane & 7;
            for (int round = 0; round < 4; ++round) {
                const int cc = round * 16 + wave * 4 + grp;   // 0..63
                const int hh = h0 - FWIN + (cc >> 3);
                const int ww = w0 - FWIN + (cc & 7);
                const bool ok = (hh >= 0 && hh < HH && ww >= 0 && ww < WW);
                const float* src = mem + (((size_t)idx * NPIX) + hh * WW + ww) * CC;
                float ssq = 0.f;
                for (int j = l8; j < CC; j += 8) {
                    const float v = ok ? src[j] : 0.0f;
                    ssq += v * v;
                    cs[cc * CPAD + j] = (bf16_t)v;
                }
                ssq += __shfl_xor(ssq, 1, 32);
                ssq += __shfl_xor(ssq, 2, 32);
                ssq += __shfl_xor(ssq, 4, 32);
                if (l8 == 0) m2s[cc] = ok ? ssq : BIGF;
            }
        }
        __syncthreads();

        // 16x16 dot tile for this wave's candidate tile, K walked 32 at a time
        v8f acc = {};
        const bf16_t* abase = &qs[cl * CPAD];
        const bf16_t* bbase = &cs[cid * CPAD];
        const int asel = (lane < 16) ? 0 : 8;    // A: K{0..7,16..23} vs {8..15,24..31}
        const int bsel = (lane < 16) ? 0 : 8;    // B: K{0..7,16..23} vs {8..15,24..31}
        for (int ks = 0; ks < CC; ks += 32) {
            const v8bf alo = *(const v8bf*)(abase + ks + asel);
            const v8bf ahi = *(const v8bf*)(abase + ks + asel + 16);
            const v8bf blo = *(const v8bf*)(bbase + ks + bsel);
            const v8bf bhi = *(const v8bf*)(bbase + ks + bsel + 16);
            const v16bf a = __builtin_shufflevector(alo, ahi,
                0, 1, 2, 3, 4, 5, 6, 7, 8, 9, 10, 11, 12, 13, 14, 15);
            const v16bf bm = __builtin_shufflevector(blo, bhi,
                0, 1, 2, 3, 4, 5, 6, 7, 8, 9, 10, 11, 12, 13, 14, 15);
            acc = __builtin_amdgcn_wmma_f32_16x16x32_bf16(
                false, a, false, bm, (short)0, acc, false, false);
        }

        // d2 = e2 + m2 - 2*dot, window/bounds mask, running min
        const float m2c = m2s[cid];
        for (int r = 0; r < 8; ++r) {
            const int q  = r + ((lane < 16) ? 0 : 8);
            const int qh = h0 + (q >> 2), qw = w0 + (q & 3);
            int dh = chh - qh; if (dh < 0) dh = -dh;
            int dw = cww - qw; if (dw < 0) dw = -dw;
            const bool valid = cin && (dh <= FWIN) && (dw <= FWIN);
            float d2 = e2q[q] + m2c - 2.0f * acc[r];
            d2 = valid ? d2 : BIGF;
            minv[r] = fminf(minv[r], d2);
        }
    }

    // min over 16 candidate columns (independently per half-wave)
    for (int r = 0; r < 8; ++r) {
        float v = minv[r];
        v = fminf(v, __shfl_xor(v, 1, 32));
        v = fminf(v, __shfl_xor(v, 2, 32));
        v = fminf(v, __shfl_xor(v, 4, 32));
        v = fminf(v, __shfl_xor(v, 8, 32));
        minv[r] = v;
    }
    if ((lane & 15) == 0) {
        for (int r = 0; r < 8; ++r)
            wmin[wave][r + ((lane < 16) ? 0 : 8)] = minv[r];
    }
    __syncthreads();
    if (tid < 16) {
        const float v = fminf(fminf(wmin[0][tid], wmin[1][tid]),
                              fminf(wmin[2][tid], wmin[3][tid]));
        const int qh = h0 + (tid >> 2), qw = w0 + (tid & 3);
        patch[b * NPIX + qh * WW + qw] = sqrtf(fmaxf(v, 0.0f));
    }
}

// ---------------- stage 4: bilinear 56 -> 224 ------------------------------
__global__ void resize_kernel(const float* __restrict__ patch, float* __restrict__ outr) {
    const int i = blockIdx.x * 256 + threadIdx.x;
    if (i >= NOUT) return;
    const int b  = i / (OUTHW * OUTHW);
    const int r  = i % (OUTHW * OUTHW);
    const int oy = r / OUTHW, ox = r % OUTHW;
    const float sy = (oy + 0.5f) * 0.25f - 0.5f;
    const float sx = (ox + 0.5f) * 0.25f - 0.5f;
    int y0 = (int)floorf(sy); const float fy = sy - (float)y0;
    int x0 = (int)floorf(sx); const float fx = sx - (float)x0;
    int y1 = y0 + 1, x1 = x0 + 1;
    y0 = min(max(y0, 0), HH - 1); y1 = min(max(y1, 0), HH - 1);
    x0 = min(max(x0, 0), WW - 1); x1 = min(max(x1, 0), WW - 1);
    const float* p = patch + b * NPIX;
    const float v = (1.f - fy) * ((1.f - fx) * p[y0 * WW + x0] + fx * p[y0 * WW + x1])
                  +        fy  * ((1.f - fx) * p[y1 * WW + x0] + fx * p[y1 * WW + x1]);
    outr[i] = v;
}

// ---------------- stage 5: 33-tap separable Gaussian, reflect pad ----------
__device__ __forceinline__ int reflect224(int i) {
    if (i < 0) i = -i;
    if (i > OUTHW - 1) i = 2 * (OUTHW - 1) - i;
    return i;
}

__global__ void blur_kernel(const float* __restrict__ in, float* __restrict__ out,
                            int vertical) {
    __shared__ float g[33];
    if (threadIdx.x < 33) {
        const float x = (float)threadIdx.x - 16.0f;
        g[threadIdx.x] = expf(-0.5f * (x * 0.25f) * (x * 0.25f));
    }
    __syncthreads();
    float gn = 0.f;
    for (int j = 0; j < 33; ++j) gn += g[j];
    gn = 1.0f / gn;
    const int i = blockIdx.x * 256 + threadIdx.x;
    if (i >= NOUT) return;
    const int b = i / (OUTHW * OUTHW);
    const int r = i % (OUTHW * OUTHW);
    const int y = r / OUTHW, x = r % OUTHW;
    const float* p = in + b * OUTHW * OUTHW;
    float s = 0.f;
    for (int j = 0; j < 33; ++j) {
        const int t = j - 16;
        const float v = vertical ? p[reflect224(y + t) * OUTHW + x]
                                 : p[y * OUTHW + reflect224(x + t)];
        s += g[j] * v;
    }
    out[i] = s * gn;
}

// ---------------- host-side orchestration ----------------------------------
extern "C" void kernel_launch(void* const* d_in, const int* in_sizes, int n_in,
                              void* d_out, int out_size, void* d_ws, size_t ws_size,
                              hipStream_t stream) {
    const float* emb = (const float*)d_in[0];   // (2,448,56,56)
    const float* mem = (const float*)d_in[1];   // (100,56,56,448)
    float* out = (float*)d_out;                 // [pred(2) | map(2*224*224)]

    float*  wsf   = (float*)d_ws;
    float*  dotw  = wsf;                        // 200
    float*  m2n   = wsf + 200;                  // 100
    float*  e2img = wsf + 300;                  // 2
    int*    tidx  = (int*)(wsf + 304);          // 40
    float*  e2p   = wsf + 352;                  // 6272
    float*  patch = wsf + 6624;                 // 6272
    float*  e_t   = wsf + 16384;                // 2,809,856 f32
    bf16_t* e_bf  = (bf16_t*)(wsf + 2826240);   // 2,809,856 bf16
    float*  resz  = wsf + 4231168;              // 100,352
    float*  tmpb  = wsf + 4331520;              // 100,352

    zero_acc_kernel<<<1, 256, 0, stream>>>(wsf);
    transpose_kernel<<<dim3(NPIX / 16, CC / 16, 2), 256, 0, stream>>>(emb, e_t, e_bf);
    e2_kernel<<<(2 * NPIX + 255) / 256, 256, 0, stream>>>(e_t, e2p, e2img);
    imgdot_kernel<<<dim3(8, NB), 256, 0, stream>>>(mem, e_t, dotw, m2n);
    topk_kernel<<<2, 32, 0, stream>>>(dotw, m2n, e2img, out, tidx);
    winnn_kernel<<<dim3(196, 2), 128, 0, stream>>>(mem, e_bf, e2p, tidx, patch);
    resize_kernel<<<(NOUT + 255) / 256, 256, 0, stream>>>(patch, resz);
    blur_kernel<<<(NOUT + 255) / 256, 256, 0, stream>>>(resz, tmpb, 1);
    blur_kernel<<<(NOUT + 255) / 256, 256, 0, stream>>>(tmpb, out + 2, 0);
}
